// ConvLayer_3204045603657
// MI455X (gfx1250) — matrix operand order repaired and checked by reading, not compiled, not run on hardware
//
#include <hip/hip_runtime.h>
#include <hip/hip_bf16.h>
#include <math.h>

typedef __attribute__((ext_vector_type(4)))  __bf16 bf16x4;
typedef __attribute__((ext_vector_type(8)))  __bf16 bf16x8;
typedef __attribute__((ext_vector_type(16))) __bf16 bf16x16;
typedef __attribute__((ext_vector_type(8)))  float  v8f;
typedef __attribute__((ext_vector_type(4)))  float  f32x4;

#define C_CH   512
#define LLEN   4096
#define LH     2048
#define KDIM   1536      // 3 taps * 512 in-channels
#define TMAIN  2046      // valid conv outputs per parity
#define BATCH  16

// Toggle the CDNA5 async global->LDS DMA path for the (already-bf16) A tile.
#define USE_ASYNC_A 1

// ---------------------------------------------------------------------------
// 1) Repack conv weights [O=512][I=512][K=3] f32 -> bf16 [O][kk = k*512 + i]
// ---------------------------------------------------------------------------
__global__ __launch_bounds__(256)
void prep_weights(const float* __restrict__ wdown, __bf16* __restrict__ wB)
{
    int idx = blockIdx.x * 256 + threadIdx.x;     // 512*1536 total
    int o  = idx / KDIM;
    int kk = idx - o * KDIM;
    int k  = kk >> 9;
    int i  = kk & 511;
    wB[(size_t)o * KDIM + kk] = (__bf16)wdown[(size_t)o * KDIM + i * 3 + k];
}

// ---------------------------------------------------------------------------
// 2) Conv-as-GEMM with V_WMMA_F32_16X16X32_BF16, double-buffered LDS,
//    async DMA (ASYNCcnt) for the weight tile, software-pipelined K loop.
//    M = out-channels (512), K = 1536, N = positions t (per batch, per parity).
//    Block tile 128x128, 8 waves (2 x 4), wave tile 64x32 (4x2 WMMA accs).
// ---------------------------------------------------------------------------
__global__ __launch_bounds__(256)
void conv_gemm(const float* __restrict__ x, const __bf16* __restrict__ wB,
               const float* __restrict__ bdown, float* __restrict__ xp)
{
    __shared__ __align__(16) __bf16 As[2][128 * 32];   // [buf][m_local][kk_local]
    __shared__ __align__(16) __bf16 Bs[2][128 * 32];   // [buf][n_local][kk_local]

    const int tid  = threadIdx.x;
    const int lane = tid & 31;
    const int wv   = tid >> 5;
    const int wm   = wv & 1;          // 0..1  (M direction)
    const int wn   = wv >> 1;         // 0..3  (N direction)

    const int blk  = blockIdx.x;      // 4 (M) * 16 (N) * 32 (batch*parity)
    const int bp   = blk >> 6;
    const int rem  = blk & 63;
    const int mblk = rem & 3;
    const int nblk = rem >> 2;
    const int b    = bp >> 1;
    const int par  = bp & 1;

    const int tt0   = nblk * 128;     // base conv position t of this block
    const int mbase = mblk * 128;     // base output channel

    const float* xb = x + (size_t)b * LLEN * C_CH;

    // ---- staging helpers -------------------------------------------------
    // A tile: 128 rows x 32 bf16 (64 B/row) = 512 x 16B chunks, 2 per thread.
    auto issueA = [&](int ks, int buf) {
        #pragma unroll
        for (int u = 0; u < 2; ++u) {
            int q   = tid + u * 256;
            int row = q >> 2;
            int seg = q & 3;
            const __bf16* g = wB + (size_t)(mbase + row) * KDIM + ks + seg * 8;
#if USE_ASYNC_A
            unsigned lds = (unsigned)(size_t)(&As[buf][row * 32 + seg * 8]);
            unsigned long long ga = (unsigned long long)(size_t)g;
            // CDNA5 async DMA: memory -> LDS, tracked with ASYNCcnt.
            asm volatile("global_load_async_to_lds_b128 %0, %1, off"
                         :: "v"(lds), "v"(ga) : "memory");
#else
            *reinterpret_cast<uint4*>(&As[buf][row * 32 + seg * 8]) =
                *reinterpret_cast<const uint4*>(g);
#endif
        }
    };

    // B tile: Bs[n][j] = bf16(x[b, 2*(tt0+n+ktap)+par, i0+j]); 4 chunks/thread.
    f32x4 breg[4];
    auto loadB = [&](int ks) {
        const int ktap = ks >> 9;
        const int i0   = ks & 511;
        #pragma unroll
        for (int u = 0; u < 4; ++u) {
            int q  = tid + u * 256;
            int n  = q >> 3;
            int j0 = (q & 7) << 2;
            int tt = tt0 + n + ktap;
            f32x4 v = {0.f, 0.f, 0.f, 0.f};
            if (tt < LH) {
                int srow = 2 * tt + par;
                v = *reinterpret_cast<const f32x4*>(xb + (size_t)srow * C_CH + i0 + j0);
            }
            breg[u] = v;
        }
    };
    auto storeB = [&](int buf) {
        #pragma unroll
        for (int u = 0; u < 4; ++u) {
            int q  = tid + u * 256;
            int n  = q >> 3;
            int j0 = (q & 7) << 2;
            bf16x4 bv;
            bv.x = (__bf16)breg[u].x; bv.y = (__bf16)breg[u].y;
            bv.z = (__bf16)breg[u].z; bv.w = (__bf16)breg[u].w;
            *reinterpret_cast<bf16x4*>(&Bs[buf][n * 32 + j0]) = bv;
        }
    };
    // ----------------------------------------------------------------------

    v8f acc[4][2];
    #pragma unroll
    for (int i = 0; i < 4; ++i)
        #pragma unroll
        for (int j = 0; j < 2; ++j) {
            v8f z = {0.f,0.f,0.f,0.f,0.f,0.f,0.f,0.f};
            acc[i][j] = z;
        }

    // Per-lane bias for the 8 channel rows of each accumulator (C/D layout:
    // lane l, elem r -> channel (l>>4)*8 + r within the 16x16 tile).
    const int chw = mbase + wm * 64;
    float bias[4][8];
    #pragma unroll
    for (int mt = 0; mt < 4; ++mt) {
        int ch0 = chw + mt * 16 + ((lane >> 4) << 3);
        #pragma unroll
        for (int r = 0; r < 8; ++r) bias[mt][r] = bdown[ch0 + r];
    }

    // ---- prologue: stage tile 0 into buffer 0 ----
    issueA(0, 0);
    loadB(0);
    storeB(0);
#if USE_ASYNC_A
    asm volatile("s_wait_asynccnt 0x0" ::: "memory");
#endif
    __syncthreads();

    int buf = 0;
    for (int ks = 0; ks < KDIM; ks += 32) {
        const int nks = ks + 32;
        const bool has_next = (nks < KDIM);

        // Kick off next tile's data movement before touching the matrix pipe.
        if (has_next) {
            issueA(nks, buf ^ 1);     // async DMA straight into LDS
            loadB(nks);               // f32 staging regs (needs cvt)
        }

        // --- fragments + 8 WMMAs from the CURRENT buffer ---
        // A (16-bit 16x32): lanes 0-15 hold K 0-7 & 16-23; lanes 16-31: 8-15 & 24-31
        const int ka = (lane >> 4) << 3;      // 0 or 8 elements
        bf16x16 afrag[4];
        #pragma unroll
        for (int mt = 0; mt < 4; ++mt) {
            int ml = wm * 64 + mt * 16 + (lane & 15);
            bf16x8 lo = *reinterpret_cast<const bf16x8*>(&As[buf][ml * 32 + ka]);
            bf16x8 hi = *reinterpret_cast<const bf16x8*>(&As[buf][ml * 32 + ka + 16]);
            afrag[mt] = __builtin_shufflevector(lo, hi,
                0,1,2,3,4,5,6,7,8,9,10,11,12,13,14,15);
        }
        // B (32x16): lanes 0-15 hold K 0-7 & 16-23; lanes 16-31: 8-15 & 24-31
        const int kb = (lane >> 4) << 3;      // 0 or 8 elements
        #pragma unroll
        for (int nt = 0; nt < 2; ++nt) {
            int nl = wn * 32 + nt * 16 + (lane & 15);
            bf16x8 lo = *reinterpret_cast<const bf16x8*>(&Bs[buf][nl * 32 + kb]);
            bf16x8 hi = *reinterpret_cast<const bf16x8*>(&Bs[buf][nl * 32 + kb + 16]);
            bf16x16 bfrag = __builtin_shufflevector(lo, hi,
                0,1,2,3,4,5,6,7,8,9,10,11,12,13,14,15);
            #pragma unroll
            for (int mt = 0; mt < 4; ++mt) {
                acc[mt][nt] = __builtin_amdgcn_wmma_f32_16x16x32_bf16(
                    false, afrag[mt], false, bfrag,
                    (short)0, acc[mt][nt], false, false);
            }
        }

        // Convert + park next B tile, then one barrier per K-step.
        if (has_next) storeB(buf ^ 1);
#if USE_ASYNC_A
        asm volatile("s_wait_asynccnt 0x0" ::: "memory");
#endif
        __syncthreads();
        buf ^= 1;
    }

    // --- epilogue: bias add + store to x_p[b][2t+par][ch] (8 contiguous floats) ---
    const int tposb = tt0 + wn * 32;
    #pragma unroll
    for (int mt = 0; mt < 4; ++mt) {
        int ch0 = chw + mt * 16 + ((lane >> 4) << 3);
        #pragma unroll
        for (int nt = 0; nt < 2; ++nt) {
            int tpos = tposb + nt * 16 + (lane & 15);
            if (tpos < TMAIN) {
                int mo = 2 * tpos + par;
                float* dst = xp + ((size_t)b * LLEN + mo) * C_CH + ch0;
                v8f v = acc[mt][nt];
                f32x4 lo = { v[0] + bias[mt][0], v[1] + bias[mt][1],
                             v[2] + bias[mt][2], v[3] + bias[mt][3] };
                f32x4 hi = { v[4] + bias[mt][4], v[5] + bias[mt][5],
                             v[6] + bias[mt][6], v[7] + bias[mt][7] };
                *reinterpret_cast<f32x4*>(dst)     = lo;
                *reinterpret_cast<f32x4*>(dst + 4) = hi;
            }
        }
    }
}

// ---------------------------------------------------------------------------
// 3) Pad positions m = 4092..4095: x_p = b_pad + w_pad(1x1) * x[first rows]
// ---------------------------------------------------------------------------
__global__ __launch_bounds__(256)
void pad_conv(const float* __restrict__ x, const float* __restrict__ wpad,
              const float* __restrict__ bpad, float* __restrict__ xp)
{
    int m    = 4092 + (blockIdx.x & 3);
    int b    = blockIdx.x >> 2;
    int par  = m & 1;
    int j    = (m >> 1) - TMAIN;          // 0 or 1
    int srow = 2 * j + par;
    const float* xr = x + ((size_t)b * LLEN + srow) * C_CH;
    for (int c = threadIdx.x; c < C_CH; c += 256) {
        const float* wr = wpad + (size_t)c * C_CH;
        float acc = bpad[c];
        for (int i = 0; i < C_CH; ++i) acc += wr[i] * xr[i];
        xp[((size_t)b * LLEN + m) * C_CH + c] = acc;
    }
}

// ---------------------------------------------------------------------------
// 4/5) BatchNorm batch stats over (B, L) per channel, two-stage reduction
// ---------------------------------------------------------------------------
__global__ __launch_bounds__(256)
void bn_partial(const float* __restrict__ xp,
                float* __restrict__ ps, float* __restrict__ pq)
{
    int blk = blockIdx.x;                 // 1024 blocks, 64 rows each
    int tid = threadIdx.x;
    size_t r0 = (size_t)blk * 64;
    float s0 = 0.f, q0 = 0.f, s1 = 0.f, q1 = 0.f;
    for (int r = 0; r < 64; ++r) {
        const float* row = xp + (r0 + r) * C_CH;
        float a = row[tid], c = row[tid + 256];
        s0 += a; q0 += a * a;
        s1 += c; q1 += c * c;
    }
    ps[(size_t)blk * C_CH + tid]       = s0;
    ps[(size_t)blk * C_CH + tid + 256] = s1;
    pq[(size_t)blk * C_CH + tid]       = q0;
    pq[(size_t)blk * C_CH + tid + 256] = q1;
}

__global__ __launch_bounds__(256)
void bn_finalize(const float* __restrict__ ps, const float* __restrict__ pq,
                 float* __restrict__ mean, float* __restrict__ rstd)
{
    int ch = blockIdx.x * 256 + threadIdx.x;   // grid = 2 blocks
    float s = 0.f, q = 0.f;
    for (int i = 0; i < 1024; ++i) {
        s += ps[(size_t)i * C_CH + ch];
        q += pq[(size_t)i * C_CH + ch];
    }
    const float inv = 1.0f / 65536.0f;         // B*L values per channel
    float m   = s * inv;
    float var = q * inv - m * m;
    mean[ch] = m;
    rstd[ch] = rsqrtf(var + 1e-5f);
}

// ---------------------------------------------------------------------------
// 6) Fused BN + ELU + residual + MaxPool(3,2,pad=1), out [B][LH][C]
// ---------------------------------------------------------------------------
__global__ __launch_bounds__(256)
void fuse_pool(const float* __restrict__ xp, const float* __restrict__ x,
               const float* __restrict__ mean, const float* __restrict__ rstd,
               const float* __restrict__ gamma, const float* __restrict__ beta,
               float* __restrict__ out)
{
    int idx = blockIdx.x;          // BATCH * LH
    int b   = idx >> 11;
    int pl  = idx & 2047;
    for (int c = threadIdx.x; c < C_CH; c += 256) {
        float mn = mean[c], rs = rstd[c], g = gamma[c], be = beta[c];
        float best = -INFINITY;
        #pragma unroll
        for (int dm = -1; dm <= 1; ++dm) {
            int m = 2 * pl + dm;
            if (m >= 0 && m < LLEN) {
                size_t off = ((size_t)b * LLEN + m) * C_CH + c;
                float v  = xp[off];
                float xn = (v - mn) * rs * g + be;
                float e  = xn > 0.f ? xn : expm1f(xn);   // ELU(alpha=1)
                best = fmaxf(best, e + x[off]);
            }
        }
        out[((size_t)b * LH + pl) * C_CH + c] = best;
    }
}

// ---------------------------------------------------------------------------
// launch: ws layout  [0,1.5MB) wB | 2MB mean/rstd | 4MB ps | 6MB pq | 8MB x_p(128MB)
// ---------------------------------------------------------------------------
extern "C" void kernel_launch(void* const* d_in, const int* in_sizes, int n_in,
                              void* d_out, int out_size, void* d_ws, size_t ws_size,
                              hipStream_t stream)
{
    const float* x     = (const float*)d_in[0];
    const float* wdown = (const float*)d_in[1];
    const float* bdown = (const float*)d_in[2];
    const float* wpad  = (const float*)d_in[3];
    const float* bpad  = (const float*)d_in[4];
    const float* gamma = (const float*)d_in[5];
    const float* beta  = (const float*)d_in[6];
    float* out = (float*)d_out;

    char*   ws   = (char*)d_ws;
    __bf16* wB   = (__bf16*)(ws);
    float*  mean = (float*)(ws + (size_t)(2u << 20));
    float*  rstd = mean + 512;
    float*  ps   = (float*)(ws + (size_t)(4u << 20));
    float*  pq   = (float*)(ws + (size_t)(6u << 20));
    float*  xp   = (float*)(ws + (size_t)(8u << 20));   // [B][L][C] f32, 128 MB

    prep_weights<<<(512 * KDIM) / 256, 256, 0, stream>>>(wdown, wB);
    conv_gemm   <<<4 * 16 * 32,        256, 0, stream>>>(x, wB, bdown, xp);
    pad_conv    <<<BATCH * 4,          256, 0, stream>>>(x, wpad, bpad, xp);
    bn_partial  <<<1024,               256, 0, stream>>>(xp, ps, pq);
    bn_finalize <<<2,                  256, 0, stream>>>(ps, pq, mean, rstd);
    fuse_pool   <<<BATCH * LH,         256, 0, stream>>>(xp, x, mean, rstd,
                                                         gamma, beta, out);
}
